// CustomAttention_25245817765955
// MI455X (gfx1250) — matrix operand order repaired and checked by reading, not compiled, not run on hardware
//
#include <hip/hip_runtime.h>
#include <math.h>

#define EMBED    1024
#define NHEADS   16
#define HEADDIM  64
#define NQ       2048
#define NK       2048
#define NBATCH   2
#define ROWS     (NBATCH * NQ)     // 4096 flat rows
#define SCALE    0.125f            // 1/sqrt(64)

typedef __attribute__((ext_vector_type(16))) __bf16 v16bf;
typedef __attribute__((ext_vector_type(8)))  float  v8f;

__device__ __forceinline__ v8f wmma_bf16(v16bf a, v16bf b, v8f c) {
    return __builtin_amdgcn_wmma_f32_16x16x32_bf16(
        false, a, false, b, (short)0, c, false, false);
}

// 16-element bf16 fragment from two 16-byte chunks (global or LDS).
__device__ __forceinline__ v16bf load_frag16(const __bf16* p0, const __bf16* p1) {
    union { uint4 u[2]; v16bf v; } t;
    t.u[0] = *reinterpret_cast<const uint4*>(p0);
    t.u[1] = *reinterpret_cast<const uint4*>(p1);
    return t.v;
}

// Reduction within each 16-lane half of a wave32 (rows of the WMMA C tile
// live in half-waves: m = r + 8*(lane>>4)).
__device__ __forceinline__ float halfmax(float v) {
#pragma unroll
    for (int m = 1; m <= 8; m <<= 1) v = fmaxf(v, __shfl_xor(v, m, 32));
    return v;
}
__device__ __forceinline__ float halfsum(float v) {
#pragma unroll
    for (int m = 1; m <= 8; m <<= 1) v += __shfl_xor(v, m, 32);
    return v;
}

// ---------------------------------------------------------------------------
// fp32 -> bf16 elementwise (bandwidth-trivial; one thread = 8 elements).
// ---------------------------------------------------------------------------
__global__ __launch_bounds__(256) void cvt_bf16_kernel(
    const float* __restrict__ src, __bf16* __restrict__ dst, int n8) {
    const int t = blockIdx.x * blockDim.x + threadIdx.x;
    if (t >= n8) return;
    const float4* s = reinterpret_cast<const float4*>(src) + (size_t)t * 2;
    const float4 u = s[0], v = s[1];
    union { uint4 q; __bf16 b[8]; } o;
    o.b[0] = (__bf16)u.x; o.b[1] = (__bf16)u.y;
    o.b[2] = (__bf16)u.z; o.b[3] = (__bf16)u.w;
    o.b[4] = (__bf16)v.x; o.b[5] = (__bf16)v.y;
    o.b[6] = (__bf16)v.z; o.b[7] = (__bf16)v.w;
    reinterpret_cast<uint4*>(dst)[t] = o.q;
}

// ---------------------------------------------------------------------------
// Pure-bf16 projection: Y(head-major [h][row][64]) = X @ W^T + bias.
// One wave: 16-row x 64-col strip (one head).  Software-pipelined: fragments
// for k-step kk+32 are loaded before the WMMAs of kk are issued.
// ---------------------------------------------------------------------------
__global__ __launch_bounds__(256) void proj_kernel(
    const __bf16* __restrict__ X, const __bf16* __restrict__ W,
    const float* __restrict__ bias, __bf16* __restrict__ Y) {
    const int lane = threadIdx.x & 31;
    const int wave = threadIdx.x >> 5;
    const int g = lane >> 4, hm = lane & 15;

    const int sid = blockIdx.x * 8 + wave;        // 4096 strips
    const int m0 = (sid >> 4) * 16;               // row tile
    const int n0 = (sid & 15) * 64;               // head = n0/64

    v8f c0 = {}, c1 = {}, c2 = {}, c3 = {};
    const __bf16* arow = X + (size_t)(m0 + hm) * EMBED;
    const __bf16* b0r = W + (size_t)(n0 + 0 + hm) * EMBED;
    const __bf16* b1r = W + (size_t)(n0 + 16 + hm) * EMBED;
    const __bf16* b2r = W + (size_t)(n0 + 32 + hm) * EMBED;
    const __bf16* b3r = W + (size_t)(n0 + 48 + hm) * EMBED;

#define LOADF(kk, A, B0, B1, B2, B3)                                          \
    do {                                                                      \
        A  = load_frag16(arow + (kk) + g * 16, arow + (kk) + g * 16 + 8);     \
        B0 = load_frag16(b0r + (kk) + g * 16, b0r + (kk) + g * 16 + 8);       \
        B1 = load_frag16(b1r + (kk) + g * 16, b1r + (kk) + g * 16 + 8);       \
        B2 = load_frag16(b2r + (kk) + g * 16, b2r + (kk) + g * 16 + 8);       \
        B3 = load_frag16(b3r + (kk) + g * 16, b3r + (kk) + g * 16 + 8);       \
    } while (0)

    v16bf a, b0, b1, b2, b3;
    LOADF(0, a, b0, b1, b2, b3);
    for (int kk = 0; kk < EMBED - 32; kk += 32) {
        v16bf an, b0n, b1n, b2n, b3n;
        LOADF(kk + 32, an, b0n, b1n, b2n, b3n);   // in flight during WMMAs
        c0 = wmma_bf16(a, b0, c0);
        c1 = wmma_bf16(a, b1, c1);
        c2 = wmma_bf16(a, b2, c2);
        c3 = wmma_bf16(a, b3, c3);
        a = an; b0 = b0n; b1 = b1n; b2 = b2n; b3 = b3n;
    }
    c0 = wmma_bf16(a, b0, c0);
    c1 = wmma_bf16(a, b1, c1);
    c2 = wmma_bf16(a, b2, c2);
    c3 = wmma_bf16(a, b3, c3);
#undef LOADF

    const int h = n0 >> 6;
#pragma unroll
    for (int j = 0; j < 4; ++j) {
        const v8f& c = (j == 0) ? c0 : (j == 1) ? c1 : (j == 2) ? c2 : c3;
        const int d = j * 16 + hm;                // within-head dim
        const float bv = bias[n0 + d];
#pragma unroll
        for (int r = 0; r < 8; ++r) {
            const int row = m0 + r + 8 * g;
            Y[((size_t)h * ROWS + row) * HEADDIM + d] = (__bf16)(c[r] + bv);
        }
    }
}

// ---------------------------------------------------------------------------
// Pass 1: row stats (M, Z) per (b,h,q).  Block = (b, h, 128 q-rows);
// 8 waves x 16-row q-tiles.  K staged through double-buffered LDS
// (2 x 16 KB) with register staging: one barrier per 128-row chunk, and the
// next chunk's global loads overlap the current chunk's 16 WMMAs.
// ---------------------------------------------------------------------------
__global__ __launch_bounds__(256) void pass1_kernel(
    const __bf16* __restrict__ Qh, const __bf16* __restrict__ Kh,
    float* __restrict__ Mrow, float* __restrict__ Zrow) {
    __shared__ uint4 kbuf4[2][1024];              // 2 x 16 KB

    const int lane = threadIdx.x & 31;
    const int wave = threadIdx.x >> 5;
    const int g = lane >> 4, hm = lane & 15;

    const int x = blockIdx.x;                     // ((b*16 + h)*16 + qb)
    const int b = x >> 8;
    const int h = (x >> 4) & 15;
    const int q0 = (x & 15) * 128 + wave * 16;

    const __bf16* qrow = Qh + ((size_t)h * ROWS + b * NQ + q0 + hm) * HEADDIM;
    const v16bf a0 = load_frag16(qrow + g * 16, qrow + g * 16 + 8);
    const v16bf a1 = load_frag16(qrow + 32 + g * 16, qrow + 32 + g * 16 + 8);

    float rm[8], rs[8];
#pragma unroll
    for (int r = 0; r < 8; ++r) { rm[r] = -3.0e38f; rs[r] = 0.0f; }

    const __bf16* kslice = Kh + ((size_t)h * ROWS + b * NK) * HEADDIM;
    uint4 stage[4];
    {
        const uint4* src = (const uint4*)kslice;
#pragma unroll
        for (int i = 0; i < 4; ++i) stage[i] = src[threadIdx.x + 256 * i];
    }

    for (int nc = 0; nc < NK / 128; ++nc) {
        const int cur = nc & 1;
#pragma unroll
        for (int i = 0; i < 4; ++i)
            kbuf4[cur][threadIdx.x + 256 * i] = stage[i];
        __syncthreads();
        if (nc + 1 < NK / 128) {                   // overlap next chunk loads
            const uint4* src =
                (const uint4*)(kslice + (size_t)(nc + 1) * 128 * HEADDIM);
#pragma unroll
            for (int i = 0; i < 4; ++i) stage[i] = src[threadIdx.x + 256 * i];
        }
        const __bf16* kbuf = (const __bf16*)kbuf4[cur];
#pragma unroll
        for (int nt = 0; nt < 8; ++nt) {
            const __bf16* kr = kbuf + (nt * 16 + hm) * HEADDIM;
            const v16bf bb0 = load_frag16(kr + g * 16, kr + g * 16 + 8);
            const v16bf bb1 = load_frag16(kr + 32 + g * 16, kr + 32 + g * 16 + 8);
            v8f cc = {};
            cc = wmma_bf16(a0, bb0, cc);
            cc = wmma_bf16(a1, bb1, cc);
#pragma unroll
            for (int r = 0; r < 8; ++r) {          // per-lane online update
                const float xv = cc[r] * SCALE;
                const float nm = fmaxf(rm[r], xv);
                rs[r] = rs[r] * __expf(rm[r] - nm) + __expf(xv - nm);
                rm[r] = nm;
            }
        }
    }
    // cross-lane finalize (each half-wave holds one row's 16 lane-partials)
#pragma unroll
    for (int r = 0; r < 8; ++r) {
        const float tm = halfmax(rm[r]);
        const float ts = halfsum(rs[r] * __expf(rm[r] - tm));
        if (hm == 0) {
            const int idx = (b * NHEADS + h) * NQ + (q0 + r + 8 * g);
            Mrow[idx] = tm;
            Zrow[idx] = __expf(-tm) + ts;          // softmax1 "+1"
        }
    }
}

// ---------------------------------------------------------------------------
// Pass 2: head-mean attention weights.  Block = (b, 128 q-rows, 128 n-cols);
// per head: K block double-buffered through LDS, 16 WMMAs per wave per head.
// ---------------------------------------------------------------------------
__global__ __launch_bounds__(256) void pass2_kernel(
    const __bf16* __restrict__ Qh, const __bf16* __restrict__ Kh,
    const float* __restrict__ Mrow, const float* __restrict__ Zrow,
    float* __restrict__ outw) {
    __shared__ uint4 kbuf4[2][1024];              // 2 x 16 KB

    const int lane = threadIdx.x & 31;
    const int wave = threadIdx.x >> 5;
    const int g = lane >> 4, hm = lane & 15;

    const int x = blockIdx.x;                     // ((b*16 + qb)*16 + nb)
    const int b = x >> 8;
    const int q0 = ((x >> 4) & 15) * 128 + wave * 16;
    const int n0 = (x & 15) * 128;

    v8f acc[8];
#pragma unroll
    for (int t = 0; t < 8; ++t) acc[t] = (v8f){};

    uint4 stage[4];
    {
        const uint4* src =
            (const uint4*)(Kh + ((size_t)0 * ROWS + b * NK + n0) * HEADDIM);
#pragma unroll
        for (int i = 0; i < 4; ++i) stage[i] = src[threadIdx.x + 256 * i];
    }

    for (int h = 0; h < NHEADS; ++h) {
        const int cur = h & 1;
#pragma unroll
        for (int i = 0; i < 4; ++i)
            kbuf4[cur][threadIdx.x + 256 * i] = stage[i];
        __syncthreads();
        if (h + 1 < NHEADS) {                      // overlap next head's loads
            const uint4* src = (const uint4*)(
                Kh + ((size_t)(h + 1) * ROWS + b * NK + n0) * HEADDIM);
#pragma unroll
            for (int i = 0; i < 4; ++i) stage[i] = src[threadIdx.x + 256 * i];
            __builtin_prefetch(
                Qh + ((size_t)(h + 1) * ROWS + b * NQ + q0 + hm) * HEADDIM,
                0, 1);
        }

        const __bf16* qrow =
            Qh + ((size_t)h * ROWS + b * NQ + q0 + hm) * HEADDIM;
        const v16bf a0 = load_frag16(qrow + g * 16, qrow + g * 16 + 8);
        const v16bf a1 = load_frag16(qrow + 32 + g * 16, qrow + 32 + g * 16 + 8);

        float mv[8], rz[8];
#pragma unroll
        for (int r = 0; r < 8; ++r) {
            const int idx = (b * NHEADS + h) * NQ + (q0 + r + 8 * g);
            mv[r] = Mrow[idx];
            rz[r] = 1.0f / Zrow[idx];
        }
        const __bf16* kbuf = (const __bf16*)kbuf4[cur];
#pragma unroll
        for (int nt = 0; nt < 8; ++nt) {
            const __bf16* kr = kbuf + (nt * 16 + hm) * HEADDIM;
            const v16bf bb0 = load_frag16(kr + g * 16, kr + g * 16 + 8);
            const v16bf bb1 = load_frag16(kr + 32 + g * 16, kr + 32 + g * 16 + 8);
            v8f cc = {};
            cc = wmma_bf16(a0, bb0, cc);
            cc = wmma_bf16(a1, bb1, cc);
#pragma unroll
            for (int r = 0; r < 8; ++r)
                acc[nt][r] += __expf(cc[r] * SCALE - mv[r]) * rz[r];
        }
    }
#pragma unroll
    for (int nt = 0; nt < 8; ++nt)
#pragma unroll
        for (int r = 0; r < 8; ++r) {
            const size_t row = (size_t)(b * NQ + q0 + r + 8 * g);
            outw[row * NK + n0 + nt * 16 + hm] = acc[nt][r] * (1.0f / NHEADS);
        }
}

// ---------------------------------------------------------------------------
// a0[b,q] = mean_h exp(-M)/Z
// ---------------------------------------------------------------------------
__global__ __launch_bounds__(256) void a0_kernel(
    const float* __restrict__ Mrow, const float* __restrict__ Zrow,
    float* __restrict__ out) {
    const int t = blockIdx.x * blockDim.x + threadIdx.x;
    if (t >= NBATCH * NQ) return;
    const int b = t / NQ, q = t % NQ;
    float s = 0.0f;
#pragma unroll
    for (int h = 0; h < NHEADS; ++h) {
        const int idx = (b * NHEADS + h) * NQ + q;
        s += __expf(-Mrow[idx]) / Zrow[idx];
    }
    out[t] = s * (1.0f / NHEADS);
}

extern "C" void kernel_launch(void* const* d_in, const int* in_sizes, int n_in,
                              void* d_out, int out_size, void* d_ws, size_t ws_size,
                              hipStream_t stream) {
    const float* query = (const float*)d_in[0];
    const float* key   = (const float*)d_in[1];
    const float* Wq    = (const float*)d_in[2];
    const float* bq    = (const float*)d_in[3];
    const float* Wk    = (const float*)d_in[4];
    const float* bk    = (const float*)d_in[5];

    float* out_a0 = (float*)d_out;                 // [2,2048]
    float* out_w  = (float*)d_out + NBATCH * NQ;   // [2,2048,2048]

    char* ws = (char*)d_ws;
    const size_t MB = (size_t)1 << 20;
    __bf16* Xq = (__bf16*)(ws + 0 * MB);           // 8 MB  query bf16
    __bf16* Xk = (__bf16*)(ws + 8 * MB);           // 8 MB  key bf16
    __bf16* Wqb = (__bf16*)(ws + 16 * MB);         // 2 MB
    __bf16* Wkb = (__bf16*)(ws + 18 * MB);         // 2 MB
    __bf16* Qh = (__bf16*)(ws + 20 * MB);          // 8 MB  head-major Q
    __bf16* Kh = (__bf16*)(ws + 28 * MB);          // 8 MB  head-major K
    float* Mrow = (float*)(ws + 36 * MB);          // 256 KB
    float* Zrow = Mrow + NBATCH * NHEADS * NQ;     // 256 KB

    const int nX8 = ROWS * EMBED / 8;              // 524288
    const int nW8 = EMBED * EMBED / 8;             // 131072
    cvt_bf16_kernel<<<(nX8 + 255) / 256, 256, 0, stream>>>(query, Xq, nX8);
    cvt_bf16_kernel<<<(nX8 + 255) / 256, 256, 0, stream>>>(key, Xk, nX8);
    cvt_bf16_kernel<<<(nW8 + 255) / 256, 256, 0, stream>>>(Wq, Wqb, nW8);
    cvt_bf16_kernel<<<(nW8 + 255) / 256, 256, 0, stream>>>(Wk, Wkb, nW8);

    proj_kernel<<<512, 256, 0, stream>>>(Xq, Wqb, bq, Qh);   // 4096 strips
    proj_kernel<<<512, 256, 0, stream>>>(Xk, Wkb, bk, Kh);

    pass1_kernel<<<512, 256, 0, stream>>>(Qh, Kh, Mrow, Zrow);
    pass2_kernel<<<512, 256, 0, stream>>>(Qh, Kh, Mrow, Zrow, out_w);
    a0_kernel<<<(NBATCH * NQ + 255) / 256, 256, 0, stream>>>(Mrow, Zrow, out_a0);
}
